// MoELayer_46462956208689
// MI455X (gfx1250) — matrix operand order repaired and checked by reading, not compiled, not run on hardware
//
#include <hip/hip_runtime.h>

// ---------------- problem constants ----------------
#define NTOK   4096          // B*T
#define DMODEL 1024
#define FDIM   4096
#define NEXP   8
#define NASSN  (NTOK * 2)    // top-2 assignments

// ---------------- vector types (trivial, union-safe) ----------------
typedef __attribute__((ext_vector_type(16))) __bf16   v16bf;
typedef __attribute__((ext_vector_type(8)))  float    v8f;
typedef __attribute__((ext_vector_type(4)))  float    fvec4;
typedef __attribute__((ext_vector_type(4)))  unsigned uvec4;
typedef __attribute__((ext_vector_type(2)))  unsigned uvec2;

union AFrag { uvec4 q[2]; v16bf v; };

// fp32 -> bf16, round to nearest even
__device__ __forceinline__ unsigned short f2bf(float f) {
  unsigned u = __float_as_uint(f);
  unsigned r = ((u >> 16) & 1u) + 0x7FFFu;
  return (unsigned short)((u + r) >> 16);
}

// A fragment (16x32 bf16): lane m = lane%16; K chunks {klo..+7} and {klo+16..+23}, klo = lane<16 ? 0 : 8
__device__ __forceinline__ v16bf load_a_frag(const unsigned short* As, int mrow, int lane) {
  int klo = (lane & 16) ? 8 : 0;
  const uvec4* p = reinterpret_cast<const uvec4*>(As + mrow * 32 + klo);
  AFrag f; f.q[0] = p[0]; f.q[1] = p[2];   // +32 bytes = +16 elements
  return f.v;
}

// B fragment (32x16 bf16): lane n = lane%16; K chunks {kb..+7} and {kb+16..+23}, kb = lane<16 ? 0 : 8
// LDS B tile stored [n][k], k contiguous (32 per n)
__device__ __forceinline__ v16bf load_b_frag(const unsigned short* Bs, int ncol, int lane) {
  int kb = (lane & 16) ? 8 : 0;
  const uvec4* p = reinterpret_cast<const uvec4*>(Bs + ncol * 32 + kb);
  AFrag f; f.q[0] = p[0]; f.q[1] = p[2];
  return f.v;
}

// async raw 16B copy global -> LDS (CDNA5 GLOBAL_LOAD_ASYNC_TO_LDS_B128, ASYNCcnt-tracked)
__device__ __forceinline__ void async_ld16(void* lds_ptr, const void* gptr) {
  unsigned short __attribute__((address_space(3)))* lp =
      (unsigned short __attribute__((address_space(3)))*)lds_ptr;
  asm volatile("global_load_async_to_lds_b128 %0, %1, off"
               :: "v"(lp), "v"(gptr) : "memory");
}

// ---------------- kernel 1: zero expert counters ----------------
__global__ void moe_init_kernel(int* cnt) {
  if (threadIdx.x < NEXP) cnt[threadIdx.x] = 0;
}

// ---------------- kernel 2: gate (logits, top-2, softmax, counts) ----------------
__global__ __launch_bounds__(256)
void moe_gate_kernel(const float* __restrict__ x, const float* __restrict__ Wg,
                     int* __restrict__ cnt, int* __restrict__ sel, float* __restrict__ gw) {
  __shared__ __align__(16) float sWg[NEXP * DMODEL];   // 32 KB
  const int tid = threadIdx.x;
  for (int i = tid; i < (NEXP * DMODEL) / 4; i += 256)
    reinterpret_cast<fvec4*>(sWg)[i] = reinterpret_cast<const fvec4*>(Wg)[i];
  __syncthreads();

  const int n = blockIdx.x * 256 + tid;
  float acc[NEXP];
#pragma unroll
  for (int e = 0; e < NEXP; ++e) acc[e] = 0.f;

  const fvec4* xr = reinterpret_cast<const fvec4*>(x + (size_t)n * DMODEL);
  for (int k4 = 0; k4 < DMODEL / 4; ++k4) {
    fvec4 xv = xr[k4];
#pragma unroll
    for (int e = 0; e < NEXP; ++e) {
      fvec4 wv = reinterpret_cast<const fvec4*>(sWg)[e * (DMODEL / 4) + k4];
      acc[e] += xv.x * wv.x + xv.y * wv.y + xv.z * wv.z + xv.w * wv.w;
    }
  }
  // top-2 (stable: ties -> lower index, matching jax.lax.top_k)
  int e0 = 0; float s0 = acc[0];
#pragma unroll
  for (int e = 1; e < NEXP; ++e) if (acc[e] > s0) { s0 = acc[e]; e0 = e; }
  int e1 = -1; float s1 = -3.4e38f;
#pragma unroll
  for (int e = 0; e < NEXP; ++e) if (e != e0 && acc[e] > s1) { s1 = acc[e]; e1 = e; }

  float p  = __expf(s1 - s0);          // s1 <= s0, safe
  float w0 = 1.f / (1.f + p);
  float w1 = p * w0;

  sel[2 * n]     = e0;  sel[2 * n + 1] = e1;
  gw[2 * n]      = w0;  gw[2 * n + 1]  = w1;
  atomicAdd(&cnt[e0], 1);
  atomicAdd(&cnt[e1], 1);
}

// ---------------- kernel 3: exclusive scan over 8 counts ----------------
__global__ void moe_scan_kernel(const int* __restrict__ cnt, int* __restrict__ offs, int* __restrict__ cur) {
  if (threadIdx.x == 0) {
    int run = 0;
    for (int e = 0; e < NEXP; ++e) { offs[e] = run; cur[e] = run; run += cnt[e]; }
  }
}

// ---------------- kernel 4: scatter assignments into per-expert lists ----------------
__global__ void moe_scatter_kernel(const int* __restrict__ sel, int* __restrict__ cur, int* __restrict__ perm) {
  int p = blockIdx.x * 256 + threadIdx.x;     // 0..NASSN-1
  int e = sel[p];
  int r = atomicAdd(&cur[e], 1);
  perm[r] = p;
}

// ---------------- kernel 5: grouped GEMM up-proj + SiLU (bf16 WMMA, pipelined) ----------------
// grid: (FDIM/128, NASSN/128, NEXP); block: 256 (8 waves)
__global__ __launch_bounds__(256)
void moe_up_kernel(const float* __restrict__ x, const float* __restrict__ W1,
                   const int* __restrict__ cnt, const int* __restrict__ offs,
                   const int* __restrict__ perm, unsigned short* __restrict__ act) {
  __shared__ __align__(16) unsigned short As[128 * 32];
  __shared__ __align__(16) unsigned short Bs[128 * 32];   // [n][k]

  const int e  = blockIdx.z;
  const int ce = cnt[e];
  const int m0 = blockIdx.y * 128;
  if (m0 >= ce) return;
  const int n0   = blockIdx.x * 128;
  const int base = offs[e];
  const int tid  = threadIdx.x;
  const int lane = tid & 31, wave = tid >> 5;
  const int wm   = (wave & 3) * 32;   // rows  [wm, wm+32)
  const int wn   = (wave >> 2) * 64;  // cols  [wn, wn+64)

  // per-thread A chunk descriptors (4 chunks; row clamped -> unconditional global loads)
  const float* xrow[4];
  int arow[4], kq4[4];
#pragma unroll
  for (int i = 0; i < 4; ++i) {
    int c   = tid + i * 256;            // 1024 chunks: row = c/8, kq = (c%8)*4
    arow[i] = c >> 3;
    kq4[i]  = (c & 7) * 4;
    int rr  = m0 + arow[i];
    if (rr >= ce) rr = ce - 1;          // clamp: padded rows compute garbage, never stored
    xrow[i] = x + (size_t)(perm[base + rr] >> 1) * DMODEL;
  }
  // per-thread B chunk descriptors
  int bk[4], bn4[4];
#pragma unroll
  for (int i = 0; i < 4; ++i) {
    int c  = tid + i * 256;             // 1024 chunks: k = c/32, n = (c%32)*4
    bk[i]  = c >> 5;
    bn4[i] = (c & 31) * 4;
  }
  const float* wbase = W1 + (size_t)e * DMODEL * FDIM + n0;

  // prologue: stage tile 0 into registers
  fvec4 ra[4], rb[4];
#pragma unroll
  for (int i = 0; i < 4; ++i) ra[i] = *reinterpret_cast<const fvec4*>(xrow[i] + kq4[i]);
#pragma unroll
  for (int i = 0; i < 4; ++i) rb[i] = *reinterpret_cast<const fvec4*>(wbase + (size_t)bk[i] * FDIM + bn4[i]);

  v8f acc[2][4] = {};

  for (int kk = 0; kk < DMODEL; kk += 32) {
    // --- regs -> LDS (cvt fp32->bf16) ---
#pragma unroll
    for (int i = 0; i < 4; ++i) {
      uvec2 pk;
      pk.x = (unsigned)f2bf(ra[i].x) | ((unsigned)f2bf(ra[i].y) << 16);
      pk.y = (unsigned)f2bf(ra[i].z) | ((unsigned)f2bf(ra[i].w) << 16);
      *reinterpret_cast<uvec2*>(&As[arow[i] * 32 + kq4[i]]) = pk;
    }
#pragma unroll
    for (int i = 0; i < 4; ++i) {
      Bs[(bn4[i] + 0) * 32 + bk[i]] = f2bf(rb[i].x);
      Bs[(bn4[i] + 1) * 32 + bk[i]] = f2bf(rb[i].y);
      Bs[(bn4[i] + 2) * 32 + bk[i]] = f2bf(rb[i].z);
      Bs[(bn4[i] + 3) * 32 + bk[i]] = f2bf(rb[i].w);
    }
    // --- issue next tile's global loads; they complete under the WMMAs below ---
    const bool next = (kk + 32) < DMODEL;
    if (next) {
#pragma unroll
      for (int i = 0; i < 4; ++i)
        ra[i] = *reinterpret_cast<const fvec4*>(xrow[i] + kk + 32 + kq4[i]);
#pragma unroll
      for (int i = 0; i < 4; ++i)
        rb[i] = *reinterpret_cast<const fvec4*>(wbase + (size_t)(kk + 32 + bk[i]) * FDIM + bn4[i]);
    }
    if (kk + 64 < DMODEL)
      __builtin_prefetch(wbase + (size_t)(kk + 64 + bk[0]) * FDIM + bn4[0], 0, 3);
    __syncthreads();

    v16bf af[2], bfr[4];
#pragma unroll
    for (int a = 0; a < 2; ++a) af[a]  = load_a_frag(As, wm + a * 16 + (lane & 15), lane);
#pragma unroll
    for (int b = 0; b < 4; ++b) bfr[b] = load_b_frag(Bs, wn + b * 16 + (lane & 15), lane);
#pragma unroll
    for (int a = 0; a < 2; ++a)
#pragma unroll
      for (int b = 0; b < 4; ++b)
        acc[a][b] = __builtin_amdgcn_wmma_f32_16x16x32_bf16(
            false, af[a], false, bfr[b], (short)0, acc[a][b], false, false);
    __syncthreads();
  }

  // --- SiLU + store bf16 activations ---
  const int mhalf = (lane & 16) ? 8 : 0;
#pragma unroll
  for (int a = 0; a < 2; ++a) {
    int mb = wm + a * 16 + mhalf;
#pragma unroll
    for (int i = 0; i < 8; ++i) {
      int m = m0 + mb + i;
      if (m >= ce) continue;
      size_t R = (size_t)(base + m);
#pragma unroll
      for (int b = 0; b < 4; ++b) {
        float v = acc[a][b][i];
        float s = v / (1.f + __expf(-v));
        act[R * FDIM + n0 + wn + b * 16 + (lane & 15)] = f2bf(s);
      }
    }
  }
}

// ---------------- kernel 6: grouped GEMM down-proj (bf16 WMMA + async A tiles) ----------------
// grid: (DMODEL/128, NASSN/128, NEXP); block: 256
__global__ __launch_bounds__(256)
void moe_down_kernel(const unsigned short* __restrict__ act, const float* __restrict__ W2,
                     const int* __restrict__ cnt, const int* __restrict__ offs,
                     const int* __restrict__ perm, float* __restrict__ stage) {
  __shared__ __align__(16) unsigned short As[2][128 * 32];  // double-buffered (async target)
  __shared__ __align__(16) unsigned short Bs[128 * 32];

  const int e  = blockIdx.z;
  const int ce = cnt[e];
  const int m0 = blockIdx.y * 128;
  if (m0 >= ce) return;
  const int n0   = blockIdx.x * 128;
  const int base = offs[e];
  const int tid  = threadIdx.x;
  const int lane = tid & 31, wave = tid >> 5;
  const int wm = (wave & 3) * 32, wn = (wave >> 2) * 64;

  // A async-copy descriptors: 2x 16B chunks per thread, rows clamped
  const unsigned short* agsrc[2];
  int ar[2], aq[2];
#pragma unroll
  for (int i = 0; i < 2; ++i) {
    int c  = tid + i * 256;              // 512 chunks: row = c/4, q = (c%4)*8 bf16
    ar[i]  = c >> 2;
    aq[i]  = (c & 3) * 8;
    int rr = m0 + ar[i];
    if (rr >= ce) rr = ce - 1;
    agsrc[i] = act + (size_t)(base + rr) * FDIM + aq[i];
  }
  // B descriptors
  int bk[4], bn4[4];
#pragma unroll
  for (int i = 0; i < 4; ++i) {
    int c  = tid + i * 256;
    bk[i]  = c >> 5;
    bn4[i] = (c & 31) * 4;
  }
  const float* wbase = W2 + (size_t)e * FDIM * DMODEL + n0;

  // prologue: async A tile 0 -> As[0]; B tile 0 -> regs
#pragma unroll
  for (int i = 0; i < 2; ++i) async_ld16(&As[0][ar[i] * 32 + aq[i]], agsrc[i]);
  fvec4 rb[4];
#pragma unroll
  for (int i = 0; i < 4; ++i) rb[i] = *reinterpret_cast<const fvec4*>(wbase + (size_t)bk[i] * DMODEL + bn4[i]);

  v8f acc[2][4] = {};

  int buf = 0;
  for (int kk = 0; kk < FDIM; kk += 32, buf ^= 1) {
    const bool next = (kk + 32) < FDIM;
    // issue async copy of NEXT A tile into the other buffer (overlaps this tile's WMMAs)
    if (next) {
#pragma unroll
      for (int i = 0; i < 2; ++i)
        async_ld16(&As[buf ^ 1][ar[i] * 32 + aq[i]], agsrc[i] + kk + 32);
    }
    // B regs (tile kk) -> LDS with cvt
#pragma unroll
    for (int i = 0; i < 4; ++i) {
      Bs[(bn4[i] + 0) * 32 + bk[i]] = f2bf(rb[i].x);
      Bs[(bn4[i] + 1) * 32 + bk[i]] = f2bf(rb[i].y);
      Bs[(bn4[i] + 2) * 32 + bk[i]] = f2bf(rb[i].z);
      Bs[(bn4[i] + 3) * 32 + bk[i]] = f2bf(rb[i].w);
    }
    // issue next B tile's global loads
    if (next) {
#pragma unroll
      for (int i = 0; i < 4; ++i)
        rb[i] = *reinterpret_cast<const fvec4*>(wbase + (size_t)(kk + 32 + bk[i]) * DMODEL + bn4[i]);
      asm volatile("s_wait_asynccnt 0x2" ::: "memory");   // retire tile kk, keep kk+32 in flight
    } else {
      asm volatile("s_wait_asynccnt 0x0" ::: "memory");
    }
    if (kk + 64 < FDIM)
      __builtin_prefetch(wbase + (size_t)(kk + 64 + bk[0]) * DMODEL + bn4[0], 0, 3);
    __syncthreads();

    v16bf af[2], bfr[4];
#pragma unroll
    for (int a = 0; a < 2; ++a) af[a]  = load_a_frag(&As[buf][0], wm + a * 16 + (lane & 15), lane);
#pragma unroll
    for (int b = 0; b < 4; ++b) bfr[b] = load_b_frag(Bs, wn + b * 16 + (lane & 15), lane);
#pragma unroll
    for (int a = 0; a < 2; ++a)
#pragma unroll
      for (int b = 0; b < 4; ++b)
        acc[a][b] = __builtin_amdgcn_wmma_f32_16x16x32_bf16(
            false, af[a], false, bfr[b], (short)0, acc[a][b], false, false);
    __syncthreads();
  }

  // --- scatter fp32 rows into per-assignment staging buffer ---
  const int mhalf = (lane & 16) ? 8 : 0;
#pragma unroll
  for (int a = 0; a < 2; ++a) {
    int mb = wm + a * 16 + mhalf;
#pragma unroll
    for (int i = 0; i < 8; ++i) {
      int m = m0 + mb + i;
      if (m >= ce) continue;
      int p = perm[base + m];
#pragma unroll
      for (int b = 0; b < 4; ++b)
        stage[(size_t)p * DMODEL + n0 + wn + b * 16 + (lane & 15)] = acc[a][b][i];
    }
  }
}

// ---------------- kernel 7: weighted top-2 combine ----------------
__global__ __launch_bounds__(256)
void moe_combine_kernel(const float* __restrict__ stage, const float* __restrict__ gw,
                        float* __restrict__ y) {
  int id = blockIdx.x * 256 + threadIdx.x;      // over NTOK * DMODEL / 4
  int n  = id >> 8;                             // DMODEL/4 = 256 chunks per token
  int c  = (id & 255) * 4;
  float w0 = gw[2 * n], w1 = gw[2 * n + 1];
  fvec4 f0 = *reinterpret_cast<const fvec4*>(stage + (size_t)(2 * n) * DMODEL + c);
  fvec4 f1 = *reinterpret_cast<const fvec4*>(stage + (size_t)(2 * n + 1) * DMODEL + c);
  fvec4 r;
  r.x = w0 * f0.x + w1 * f1.x;
  r.y = w0 * f0.y + w1 * f1.y;
  r.z = w0 * f0.z + w1 * f1.z;
  r.w = w0 * f0.w + w1 * f1.w;
  *reinterpret_cast<fvec4*>(y + (size_t)n * DMODEL + c) = r;
}

// ---------------- launch ----------------
extern "C" void kernel_launch(void* const* d_in, const int* in_sizes, int n_in,
                              void* d_out, int out_size, void* d_ws, size_t ws_size,
                              hipStream_t stream) {
  const float* x  = (const float*)d_in[0];
  const float* Wg = (const float*)d_in[1];
  const float* W1 = (const float*)d_in[2];
  const float* W2 = (const float*)d_in[3];
  float* y = (float*)d_out;

  char* ws = (char*)d_ws;
  int*   cnt  = (int*)(ws + 0);
  int*   offs = (int*)(ws + 32);
  int*   cur  = (int*)(ws + 64);
  int*   perm = (int*)(ws + 256);                               // 8192 ints
  int*   sel  = (int*)(ws + 256 + 32768);                       // 8192 ints
  float* gw   = (float*)(ws + 256 + 65536);                     // 8192 floats
  unsigned short* act = (unsigned short*)(ws + 98560);          // 8192*4096 bf16 = 67.1 MB
  float* stage = (float*)(ws + 98560 + (size_t)NASSN * FDIM * 2); // 8192*1024 fp32 = 33.5 MB

  moe_init_kernel<<<1, 32, 0, stream>>>(cnt);
  moe_gate_kernel<<<NTOK / 256, 256, 0, stream>>>(x, Wg, cnt, sel, gw);
  moe_scan_kernel<<<1, 32, 0, stream>>>(cnt, offs, cur);
  moe_scatter_kernel<<<NASSN / 256, 256, 0, stream>>>(sel, cur, perm);
  moe_up_kernel<<<dim3(FDIM / 128, NASSN / 128, NEXP), 256, 0, stream>>>(x, W1, cnt, offs, perm, act);
  moe_down_kernel<<<dim3(DMODEL / 128, NASSN / 128, NEXP), 256, 0, stream>>>(act, W2, cnt, offs, perm, stage);
  moe_combine_kernel<<<(NTOK * DMODEL / 4) / 256, 256, 0, stream>>>(stage, gw, y);
}
